// RGINConv_6932077216184
// MI455X (gfx1250) — compile-verified, matrix operand order checked by reading, not run on hardware
//
#include <hip/hip_runtime.h>
#include <hip/hip_bf16.h>

// ---------------------------------------------------------------------------
// RGINConv for MI455X (gfx1250, wave32, WMMA + async-to-LDS weight staging).
//
//   K0 prep   : fp32 weights -> fragment-major bf16 (WMMA B-layout); zero stats
//   K1 init   : agg[r][n][:] = x[n][:]
//   K2 scatter: wave-per-edge float4 gather + 4 L2-resident float atomics/lane
//   K3 gemm1  : 8 waves/block, W1[r] staged to LDS via global_load_async_to_lds,
//               per-wave 16x128 tile, 32x v_wmma_f32_16x16x32_bf16, in-place
//               writeback + BN partial stats
//   K4 bnfin  : scale/shift from batch stats
//   K5 gemm2  : 160KB LDS staging (Wself + 4xW2), fused BN+ReLU into bf16
//               A-fragments, 160 WMMAs per tile, direct output writeback
// ---------------------------------------------------------------------------

typedef __attribute__((ext_vector_type(16))) __bf16 v16bf;
typedef __attribute__((ext_vector_type(8)))  float  v8f;

#define N_NODES 50000
#define N_EDGES 800000
#define F_DIM   128
#define N_REL   4
#define TILES_PER_REL 3125          // 50000 / 16, exact
#define BLK_PER_REL   391           // ceil(3125 / 8)

__device__ __forceinline__ __bf16 f2bf(float f) {   // round-to-nearest-even
    unsigned u = __builtin_bit_cast(unsigned, f);
    u += 0x7FFFu + ((u >> 16) & 1u);
    unsigned short h = (unsigned short)(u >> 16);
    return __builtin_bit_cast(__bf16, h);
}

struct F16x { float v[16]; };

// 16 fp32 values a lane contributes to a 16x32 bf16 A fragment.
// lane<16 -> K = kb + {0..7,16..23}; lane>=16 -> kb += 8 (caller folds hi*8).
__device__ __forceinline__ F16x load16(const float* rp, int kb) {
    const float4* p = (const float4*)(rp + kb);
    float4 q0 = p[0], q1 = p[1], q2 = p[4], q3 = p[5];
    F16x o;
    o.v[0]=q0.x;  o.v[1]=q0.y;  o.v[2]=q0.z;  o.v[3]=q0.w;
    o.v[4]=q1.x;  o.v[5]=q1.y;  o.v[6]=q1.z;  o.v[7]=q1.w;
    o.v[8]=q2.x;  o.v[9]=q2.y;  o.v[10]=q2.z; o.v[11]=q2.w;
    o.v[12]=q3.x; o.v[13]=q3.y; o.v[14]=q3.z; o.v[15]=q3.w;
    return o;
}

__device__ __forceinline__ v16bf pack_bf(const F16x& f) {
    v16bf a;
#pragma unroll
    for (int i = 0; i < 16; ++i) a[i] = f2bf(f.v[i]);
    return a;
}

// Async copy of one 16B chunk global -> LDS (CDNA5 async path, ASYNCcnt).
__device__ __forceinline__ void async_cp16(unsigned lds_off, const char* gaddr) {
    unsigned long long ga = (unsigned long long)gaddr;
    asm volatile("global_load_async_to_lds_b128 %0, %1, off"
                 :: "v"(lds_off), "v"(ga) : "memory");
}
__device__ __forceinline__ void async_wait0() {
    asm volatile("s_wait_asynccnt 0x0" ::: "memory");
}

// --------------------------- K0: weight prep --------------------------------
// matrices m: {W1 r0..3, W2 r0..3, Wself}; frag = nt*4 + kc; lane owns 32B.
__global__ void k_prep(const float* __restrict__ W1, const float* __restrict__ W2,
                       const float* __restrict__ Wself, __bf16* __restrict__ wbf,
                       float* __restrict__ stats /* 1024 floats: sum|sumsq */) {
    int t = blockIdx.x * 256 + threadIdx.x;          // 147456 threads exactly
    if (t < 1024) stats[t] = 0.0f;                   // re-zeroed every launch
    int m    = t >> 14;
    int rem  = t & 16383;
    int f    = rem >> 9;                             // fragment 0..31
    int lane = (rem >> 4) & 31;
    int j    = rem & 15;
    int nt = f >> 2, kc = f & 3;
    int klocal = ((lane & 16) ? 8 : 0) + (j < 8 ? j : j + 8);
    int K  = kc * 32 + klocal;
    int Nn = nt * 16 + (lane & 15);
    const float* src = (m < 4) ? (W1 + (size_t)m * 16384)
                     : (m < 8) ? (W2 + (size_t)(m - 4) * 16384)
                               : Wself;
    wbf[(size_t)m * 16384 + (size_t)f * 512 + lane * 16 + j] = f2bf(src[K * F_DIM + Nn]);
}

// --------------------------- K1: agg init -----------------------------------
__global__ void k_init(const float* __restrict__ x, float* __restrict__ agg) {
    size_t i = (size_t)blockIdx.x * 256 + threadIdx.x;        // 6,400,000 float4s
    size_t s = i % (size_t)(N_NODES * (F_DIM / 4));
    ((float4*)agg)[i] = ((const float4*)x)[s];
}

// --------------------------- K2: edge scatter -------------------------------
__global__ void k_scatter(const float* __restrict__ x, const int* __restrict__ ei,
                          const int* __restrict__ et, float* __restrict__ agg) {
    int wave = (blockIdx.x * 256 + threadIdx.x) >> 5;         // 800,000 waves
    int lane = threadIdx.x & 31;
    if (wave >= N_EDGES) return;
    int src = ei[wave];
    int dst = ei[N_EDGES + wave];
    int r   = et[wave];
    float4 v = ((const float4*)(x + (size_t)src * F_DIM))[lane];   // coalesced
    float* dr = agg + ((size_t)r * N_NODES + dst) * F_DIM + lane * 4;
    atomicAdd(dr + 0, v.x); atomicAdd(dr + 1, v.y);           // L2-resident RMW
    atomicAdd(dr + 2, v.z); atomicAdd(dr + 3, v.w);
}

// --------------------------- K3: GEMM1 + BN stats ---------------------------
// 8 waves/block share W1[r] staged in 32KB LDS; wave -> one 16-row tile.
__global__ void __launch_bounds__(256)
k_gemm1(float* __restrict__ agg, const __bf16* __restrict__ wbf,
        const float* __restrict__ b1, float* __restrict__ ssum,
        float* __restrict__ ssq) {
    extern __shared__ char smem[];                    // 32 KB
    int tid = threadIdx.x;
    int r   = blockIdx.x / BLK_PER_REL;
    int blk = blockIdx.x % BLK_PER_REL;

    const char* gw   = (const char*)(wbf + (size_t)r * 16384);
    unsigned    lbas = (unsigned)(size_t)smem;
#pragma unroll
    for (int it = 0; it < 8; ++it)                    // 256 thr * 16B * 8 = 32KB
        async_cp16(lbas + tid * 16 + it * 4096, gw + tid * 16 + it * 4096);
    async_wait0();
    __syncthreads();

    int wave = tid >> 5, lane = tid & 31, lm = lane & 15, hi = lane >> 4;
    int tin = blk * 8 + wave;
    if (tin < TILES_PER_REL) {                        // whole-wave predicate
        int row0 = tin * 16;
        const float* rowp = agg + ((size_t)r * N_NODES + row0 + lm) * F_DIM;

        v8f acc[8];
#pragma unroll
        for (int nt = 0; nt < 8; ++nt) {
            float bv = b1[r * F_DIM + nt * 16 + lm];
#pragma unroll
            for (int v = 0; v < 8; ++v) acc[nt][v] = bv;
        }
#pragma unroll
        for (int kc = 0; kc < 4; ++kc) {
            v16bf a = pack_bf(load16(rowp, kc * 32 + hi * 8));
            v16bf bfr[8];
#pragma unroll
            for (int nt = 0; nt < 8; ++nt)            // ds_load_b128 x2 each
                bfr[nt] = *(const v16bf*)(smem + (size_t)(nt * 4 + kc) * 1024 + lane * 32);
#pragma unroll
            for (int nt = 0; nt < 8; ++nt)            // back-to-back WMMAs
                acc[nt] = __builtin_amdgcn_wmma_f32_16x16x32_bf16(
                    false, a, false, bfr[nt], (short)0, acc[nt], false, false);
        }
        // in-place writeback (lane<16 -> M=v, lane>=16 -> M=v+8; N=nt*16+lm)
#pragma unroll
        for (int nt = 0; nt < 8; ++nt) {
            float s = 0.0f, q = 0.0f;
#pragma unroll
            for (int v = 0; v < 8; ++v) {
                float h = acc[nt][v];
                agg[((size_t)r * N_NODES + row0 + v + hi * 8) * F_DIM + nt * 16 + lm] = h;
                s += h; q += h * h;
            }
            s += __shfl_xor(s, 16, 32);
            q += __shfl_xor(q, 16, 32);
            if (hi == 0) {
                atomicAdd(&ssum[r * F_DIM + nt * 16 + lm], s);
                atomicAdd(&ssq [r * F_DIM + nt * 16 + lm], q);
            }
        }
    }
}

// --------------------------- K4: BN finalize --------------------------------
__global__ void k_bnfin(const float* __restrict__ ssum, const float* __restrict__ ssq,
                        const float* __restrict__ gamma, const float* __restrict__ beta,
                        float* __restrict__ scale, float* __restrict__ shift) {
    int t = blockIdx.x * 256 + threadIdx.x;
    if (t >= N_REL * F_DIM) return;
    const float invN = 1.0f / (float)N_NODES;
    float mean = ssum[t] * invN;
    float var  = ssq[t] * invN - mean * mean;
    float inv  = rsqrtf(var + 1e-5f);
    float sc   = gamma[t] * inv;
    scale[t] = sc;
    shift[t] = beta[t] - mean * sc;
}

// --------------------------- K5: fused GEMM2 + self -------------------------
// 160KB LDS: [Wself | W2 r0..3]; 8 waves/block, tile per wave, 160 WMMAs/tile.
__global__ void __launch_bounds__(256)
k_gemm2(const float* __restrict__ x, const float* __restrict__ agg,
        const __bf16* __restrict__ wbf, const float* __restrict__ bself,
        const float* __restrict__ b2, const float* __restrict__ scale,
        const float* __restrict__ shift, float* __restrict__ out) {
    extern __shared__ char smem[];                    // 160 KB
    int tid = threadIdx.x;

    const char* gself = (const char*)(wbf + (size_t)8 * 16384);   // 32KB
    const char* gw2   = (const char*)(wbf + (size_t)4 * 16384);   // 128KB
    unsigned    lbas  = (unsigned)(size_t)smem;
#pragma unroll
    for (int it = 0; it < 8; ++it)
        async_cp16(lbas + tid * 16 + it * 4096, gself + tid * 16 + it * 4096);
#pragma unroll
    for (int it = 0; it < 32; ++it)
        async_cp16(lbas + 32768 + tid * 16 + it * 4096, gw2 + tid * 16 + it * 4096);
    async_wait0();
    __syncthreads();

    int wave = tid >> 5, lane = tid & 31, lm = lane & 15, hi = lane >> 4;
    int tile = blockIdx.x * 8 + wave;
    if (tile >= TILES_PER_REL) return;                // whole-wave exit, post-barrier
    int row0 = tile * 16;

    v8f acc[8];
#pragma unroll
    for (int nt = 0; nt < 8; ++nt) {
        int c = nt * 16 + lm;
        float bv = bself[c] + b2[c] + b2[F_DIM + c] + b2[2 * F_DIM + c] + b2[3 * F_DIM + c];
#pragma unroll
        for (int v = 0; v < 8; ++v) acc[nt][v] = bv;
    }
    // self-loop GEMM: x @ Wself  (weights at smem[0..32KB))
    {
        const float* xrow = x + (size_t)(row0 + lm) * F_DIM;
#pragma unroll
        for (int kc = 0; kc < 4; ++kc) {
            v16bf a = pack_bf(load16(xrow, kc * 32 + hi * 8));
            v16bf bfr[8];
#pragma unroll
            for (int nt = 0; nt < 8; ++nt)
                bfr[nt] = *(const v16bf*)(smem + (size_t)(nt * 4 + kc) * 1024 + lane * 32);
#pragma unroll
            for (int nt = 0; nt < 8; ++nt)
                acc[nt] = __builtin_amdgcn_wmma_f32_16x16x32_bf16(
                    false, a, false, bfr[nt], (short)0, acc[nt], false, false);
        }
    }
    // relations: relu(bn(h1)) @ W2[r]   (weights at smem[(1+r)*32KB))
    for (int r = 0; r < N_REL; ++r) {
        const float* hrow = agg + ((size_t)r * N_NODES + row0 + lm) * F_DIM;
        const float* sc   = scale + r * F_DIM;
        const float* sh   = shift + r * F_DIM;
        const char*  wl   = smem + (size_t)(1 + r) * 32768;
#pragma unroll
        for (int kc = 0; kc < 4; ++kc) {
            int kb = kc * 32 + hi * 8;
            F16x hv = load16(hrow, kb), sv = load16(sc, kb), bv = load16(sh, kb);
            v16bf a;
#pragma unroll
            for (int i = 0; i < 16; ++i)
                a[i] = f2bf(fmaxf(fmaf(hv.v[i], sv.v[i], bv.v[i]), 0.0f));
            v16bf bfr[8];
#pragma unroll
            for (int nt = 0; nt < 8; ++nt)
                bfr[nt] = *(const v16bf*)(wl + (size_t)(nt * 4 + kc) * 1024 + lane * 32);
#pragma unroll
            for (int nt = 0; nt < 8; ++nt)
                acc[nt] = __builtin_amdgcn_wmma_f32_16x16x32_bf16(
                    false, a, false, bfr[nt], (short)0, acc[nt], false, false);
        }
    }
#pragma unroll
    for (int nt = 0; nt < 8; ++nt)
#pragma unroll
        for (int v = 0; v < 8; ++v)
            out[(size_t)(row0 + v + hi * 8) * F_DIM + nt * 16 + lm] = acc[nt][v];
}

// ---------------------------------------------------------------------------
extern "C" void kernel_launch(void* const* d_in, const int* in_sizes, int n_in,
                              void* d_out, int out_size, void* d_ws, size_t ws_size,
                              hipStream_t stream) {
    const float* x     = (const float*)d_in[0];
    const int*   ei    = (const int*)d_in[1];
    const int*   et    = (const int*)d_in[2];
    const float* Wself = (const float*)d_in[3];
    const float* bself = (const float*)d_in[4];
    const float* W1    = (const float*)d_in[5];
    const float* b1    = (const float*)d_in[6];
    const float* gamma = (const float*)d_in[7];
    const float* beta  = (const float*)d_in[8];
    const float* W2    = (const float*)d_in[9];
    const float* b2    = (const float*)d_in[10];
    float* out = (float*)d_out;

    // Workspace (~102.7 MB): agg/h1 | stats | bn scale/shift | bf16 weights
    char* ws = (char*)d_ws;
    float*  agg   = (float*)ws;
    size_t  off   = (size_t)N_REL * N_NODES * F_DIM * sizeof(float); // 102,400,000
    float*  ssum  = (float*)(ws + off);
    float*  ssq   = (float*)(ws + off + 2048);
    float*  bnsc  = (float*)(ws + off + 4096);
    float*  bnsh  = (float*)(ws + off + 6144);
    __bf16* wbf   = (__bf16*)(ws + off + 8192);

    // Allow 160KB dynamic LDS for gemm2 (gfx1250 WGP has 320KB). Host-side
    // attribute set; not a stream operation, so graph-capture safe.
    hipFuncSetAttribute((const void*)k_gemm2,
                        hipFuncAttributeMaxDynamicSharedMemorySize, 163840);

    k_prep   <<<576,    256, 0, stream>>>(W1, W2, Wself, wbf, ssum);
    k_init   <<<25000,  256, 0, stream>>>(x, agg);
    k_scatter<<<100000, 256, 0, stream>>>(x, ei, et, agg);
    k_gemm1  <<<N_REL * BLK_PER_REL, 256, 32768, stream>>>(agg, wbf, b1, ssum, ssq);
    k_bnfin  <<<2,      256, 0, stream>>>(ssum, ssq, gamma, beta, bnsc, bnsh);
    k_gemm2  <<<BLK_PER_REL, 256, 163840, stream>>>(x, agg, wbf, bself, b2, bnsc, bnsh, out);
}
